// LlamaAttention_50611894616321
// MI455X (gfx1250) — matrix operand order repaired and checked by reading, not compiled, not run on hardware
//
#include <hip/hip_runtime.h>
#include <hip/hip_bf16.h>

// ---------------- constants ----------------
#define BATCH   2
#define SEQ     2048
#define CDIM    2048          // hidden = K-dim of all dense GEMMs
#define HEADS   16
#define HDIM    128
#define MTOT    (BATCH*SEQ)   // 4096 rows

// softmax: scale * log2(e)  with scale = 1/sqrt(128)
#define KLS     0.12751744f
// log2(10000)/64 for RoPE inv_freq = exp2(-dp * L2T64)
#define L2T64   0.20762050593046013f

typedef __attribute__((ext_vector_type(16))) _Float16 v16h;
typedef __attribute__((ext_vector_type(8)))  float    v8f;
typedef __attribute__((ext_vector_type(4)))  unsigned int u32x4;
typedef __attribute__((ext_vector_type(8)))  int          i32x8;
typedef __attribute__((ext_vector_type(4)))  int          i32x4;

#if defined(__has_builtin)
#  if __has_builtin(__builtin_amdgcn_tensor_load_to_lds)
#    define HAVE_TDM 1
#  else
#    define HAVE_TDM 0
#  endif
#else
#  define HAVE_TDM 0
#endif

// ---------------- fragment helpers ----------------
__device__ __forceinline__ v8f wmma32(v16h a, v16h b, v8f c) {
  return __builtin_amdgcn_wmma_f32_16x16x32_f16(false, a, false, b,
                                                (short)0, c, false, false);
}

// 8 f16 @p0 -> elems 0..7, 8 f16 @p1 -> elems 8..15
__device__ __forceinline__ v16h load_frag_h(const _Float16* p0, const _Float16* p1) {
  union { uint4 u[2]; v16h v; } r;
  r.u[0] = *reinterpret_cast<const uint4*>(p0);
  r.u[1] = *reinterpret_cast<const uint4*>(p1);
  return r.v;
}
__device__ __forceinline__ v16h load_frag_h16(const _Float16* p) {
  return load_frag_h(p, p + 8);
}

#if HAVE_TDM
// ---------------- Tensor Data Mover: 2D tile (f16) global -> LDS ----------------
// D# per CDNA5 ISA.  Tile: tile_rows x tile_k f16,
// row stride row_stride elements; packed contiguously at lds_off.
// 6-arg builtin (amdgpu-toolchain / clang-23): groups 2/3 and the trailing
// i32x8 are zero -> 2D tensor (VADDR2/3 conceptually NULL).
__device__ __forceinline__ void tdm_load_tile_2d(const _Float16* gptr,
                                                 unsigned int lds_off,
                                                 int tile_k, int tile_rows,
                                                 int tensor_k, int tensor_rows,
                                                 int row_stride) {
  const unsigned long long ga = (unsigned long long)(uintptr_t)gptr;
  u32x4 g0;
  g0[0] = 1u;                                              // count=1, user mode
  g0[1] = lds_off;                                         // lds_addr (bytes)
  g0[2] = (unsigned int)ga;                                // global_addr[31:0]
  g0[3] = (unsigned int)((ga >> 32) & 0x01FFFFFFu)         // global_addr[56:32]
        | (2u << 30);                                      // type = 2 ("image")
  i32x8 g1;
  g1[0] = (int)(1u << 16);                                 // data_size = 2 bytes
  g1[1] = (int)(((unsigned int)tensor_k & 0xFFFFu) << 16); // tensor_dim0[15:0]
  g1[2] = (int)(((unsigned int)tensor_k >> 16)
        | (((unsigned int)tensor_rows & 0xFFFFu) << 16));  // td0[31:16] | td1[15:0]
  g1[3] = (int)(((unsigned int)tensor_rows >> 16)
        | ((unsigned int)tile_k << 16));                   // td1[31:16] | tile_dim0
  g1[4] = (int)((unsigned int)tile_rows);                  // tile_dim1 (tile_dim2=0)
  g1[5] = (int)((unsigned int)row_stride);                 // tensor_dim0_stride[31:0]
  g1[6] = 0;                                               // stride hi / dim1_stride
  g1[7] = 0;
  const i32x4 z4 = {0, 0, 0, 0};
  const i32x8 z8 = {0, 0, 0, 0, 0, 0, 0, 0};
  __builtin_amdgcn_tensor_load_to_lds(g0, g1, z4, z4, z8, 0);
}
#endif

// cooperative fallback: 256 threads load 64x32 f16 tile
__device__ __forceinline__ void coop_load_tile(_Float16 (*bt)[32],
                                               const _Float16* W, int cb, int k0) {
  const int t   = threadIdx.x;
  const int row = t >> 2;
  const int ch  = (t & 3) * 8;
  *reinterpret_cast<uint4*>(&bt[row][ch]) =
      *reinterpret_cast<const uint4*>(W + (size_t)(cb + row) * CDIM + k0 + ch);
}

// ================= kernel 0: f32 -> f16 convert (x, weights) =================
__global__ __launch_bounds__(256)
void cvt_f16_kernel(const float* __restrict__ src, _Float16* __restrict__ dst, int n8) {
  const int i = blockIdx.x * 256 + threadIdx.x;
  if (i >= n8) return;
  const float4* s = reinterpret_cast<const float4*>(src) + (size_t)i * 2;
  float4 f0 = s[0], f1 = s[1];
  union { uint4 u; _Float16 h[8]; } r;
  r.h[0] = (_Float16)f0.x; r.h[1] = (_Float16)f0.y;
  r.h[2] = (_Float16)f0.z; r.h[3] = (_Float16)f0.w;
  r.h[4] = (_Float16)f1.x; r.h[5] = (_Float16)f1.y;
  r.h[6] = (_Float16)f1.z; r.h[7] = (_Float16)f1.w;
  reinterpret_cast<uint4*>(dst)[i] = r.u;
}

// ================= kernel 1: QKV GEMM (all-f16, TDM-staged B tiles) =================
// grid (MTOT/128, CDIM/64, 3), block 256.  mode 0->q, 1->k (natural [m][c]),
// mode 2->v stored transposed vt[(b*H+h)*HDIM + d][t].
__global__ __launch_bounds__(256)
void qkv_gemm_kernel(const _Float16* __restrict__ xh,
                     const _Float16* __restrict__ wq,
                     const _Float16* __restrict__ wk,
                     const _Float16* __restrict__ wv,
                     _Float16* __restrict__ qh,
                     _Float16* __restrict__ kh,
                     _Float16* __restrict__ vt)
{
  __shared__ _Float16 bt[2][64][32];   // double-buffered 32(k) x 64(n) B tile

  const int mode = blockIdx.z;
  const _Float16* __restrict__ W = (mode == 0) ? wq : (mode == 1) ? wk : wv;

  const int lane = threadIdx.x & 31;
  const int wvi  = threadIdx.x >> 5;
  const int ln   = lane & 15;
  const int lh   = lane >> 4;
  const int rb   = blockIdx.x * 128 + wvi * 16;
  const int cb   = blockIdx.y * 64;

  // ---- preload tile 0 ----
#if HAVE_TDM
  if (threadIdx.x < 32) {
    const unsigned int lds0 = (unsigned int)(uintptr_t)&bt[0][0][0];
    tdm_load_tile_2d(W + (size_t)cb * CDIM, lds0, 32, 64, CDIM, CDIM, CDIM);
    __builtin_amdgcn_s_wait_tensorcnt(0);
  }
  __syncthreads();
#else
  coop_load_tile(bt[0], W, cb, 0);
  __syncthreads();
#endif

  v8f acc[4] = {};
  const _Float16* arow = xh + (size_t)(rb + ln) * CDIM;

  for (int k0 = 0; k0 < CDIM; k0 += 32) {
    const int buf = (k0 >> 5) & 1;
    const bool more = (k0 + 32) < CDIM;
#if HAVE_TDM
    if (threadIdx.x < 32 && more) {
      const unsigned int lds1 = (unsigned int)(uintptr_t)&bt[buf ^ 1][0][0];
      tdm_load_tile_2d(W + (size_t)cb * CDIM + (k0 + 32), lds1, 32, 64,
                       CDIM, CDIM, CDIM);
    }
#else
    __syncthreads();
    if (more) coop_load_tile(bt[buf ^ 1], W, cb, k0 + 32);
#endif
    __builtin_prefetch(arow + k0 + 256, 0, 1);
    v16h a = load_frag_h16(arow + k0 + lh * 16);
#pragma unroll
    for (int nc = 0; nc < 4; ++nc) {
      v16h b = load_frag_h16(&bt[buf][nc * 16 + ln][lh * 16]);
      acc[nc] = wmma32(a, b, acc[nc]);
    }
#if HAVE_TDM
    if (threadIdx.x < 32 && more) __builtin_amdgcn_s_wait_tensorcnt(0);
#endif
    __syncthreads();
  }

  if (mode < 2) {
    _Float16* __restrict__ dst = (mode == 0) ? qh : kh;
#pragma unroll
    for (int nc = 0; nc < 4; ++nc)
#pragma unroll
      for (int j = 0; j < 8; ++j)
        dst[(size_t)(rb + j + 8 * lh) * CDIM + cb + nc * 16 + ln] =
            (_Float16)acc[nc][j];
  } else {
#pragma unroll
    for (int nc = 0; nc < 4; ++nc) {
      const int g = cb + nc * 16 + ln;
      const int h = g >> 7, d = g & 127;
#pragma unroll
      for (int j = 0; j < 8; ++j) {
        const int row = rb + j + 8 * lh;
        const int b_ = row >> 11, t = row & (SEQ - 1);
        vt[((size_t)(b_ * HEADS + h) * HDIM + d) * SEQ + t] = (_Float16)acc[nc][j];
      }
    }
  }
}

// ================= kernel 2: RoPE in-place on q,k (f16) =================
__global__ __launch_bounds__(256)
void rope_kernel(_Float16* __restrict__ qh, _Float16* __restrict__ kh)
{
  const int idx = blockIdx.x * 256 + threadIdx.x;   // [0, MTOT*HEADS*64)
  const int dp  = idx & 63;
  const int h   = (idx >> 6) & (HEADS - 1);
  const int m   = idx >> 10;
  const int t   = m & (SEQ - 1);

  const float ang = (float)t * exp2f(-L2T64 * (float)dp);
  float s, c;
  sincosf(ang, &s, &c);

  const size_t base = (size_t)m * CDIM + h * HDIM + dp;
  float q1 = (float)qh[base], q2 = (float)qh[base + 64];
  qh[base]      = (_Float16)(q1 * c - q2 * s);
  qh[base + 64] = (_Float16)(q2 * c + q1 * s);
  float k1 = (float)kh[base], k2 = (float)kh[base + 64];
  kh[base]      = (_Float16)(k1 * c - k2 * s);
  kh[base + 64] = (_Float16)(k2 * c + k1 * s);
}

// ================= kernel 3: causal flash attention =================
// grid (SEQ/128, BATCH*HEADS), block 256.  one wave = 16 query rows.
__global__ __launch_bounds__(256)
void attn_kernel(const _Float16* __restrict__ qh,
                 const _Float16* __restrict__ kh,
                 const _Float16* __restrict__ vt,
                 _Float16* __restrict__ ah)
{
  __shared__ _Float16 lp[8][16][32];   // per-wave probability tile (C->A relayout)

  const int lane = threadIdx.x & 31;
  const int wvi  = threadIdx.x >> 5;
  const int ln   = lane & 15;
  const int lh   = lane >> 4;
  const int q0   = blockIdx.x * 128 + wvi * 16;
  const int b    = blockIdx.y >> 4;
  const int h    = blockIdx.y & 15;

  v16h qa[4];
#pragma unroll
  for (int dc = 0; dc < 4; ++dc) {
    const _Float16* r = qh + (size_t)(b * SEQ + q0 + ln) * CDIM + h * HDIM + dc * 32;
    qa[dc] = load_frag_h16(r + lh * 16);
  }

  v8f  o[8] = {};
  float mrow[8], lrow[8];
#pragma unroll
  for (int j = 0; j < 8; ++j) { mrow[j] = -3.0e38f; lrow[j] = 0.0f; }

  const _Float16* kbase = kh + (size_t)(b * SEQ) * CDIM + h * HDIM;
  const _Float16* vbase = vt + (size_t)(b * HEADS + h) * HDIM * SEQ;

  const int ntiles = q0 / 32 + 1;                   // causal: last tile masked
  for (int it = 0; it < ntiles; ++it) {
    const int k0 = it * 32;

    v8f s0 = {}, s1 = {};
#pragma unroll
    for (int dc = 0; dc < 4; ++dc) {
      v16h b0 = load_frag_h16(kbase + (size_t)(k0 + ln) * CDIM + dc * 32 + lh * 16);
      s0 = wmma32(qa[dc], b0, s0);
      v16h b1 = load_frag_h16(kbase + (size_t)(k0 + 16 + ln) * CDIM + dc * 32 + lh * 16);
      s1 = wmma32(qa[dc], b1, s1);
    }

    const bool maskt = (it == ntiles - 1);
    float p0[8], p1[8];
#pragma unroll
    for (int j = 0; j < 8; ++j) {
      float a0 = s0[j], a1 = s1[j];
      if (maskt) {
        const int row = q0 + j + 8 * lh;
        if (k0 + ln > row)      a0 = -3.0e38f;
        if (k0 + 16 + ln > row) a1 = -3.0e38f;
      }
      float t = fmaxf(a0, a1);
#pragma unroll
      for (int off = 1; off < 16; off <<= 1) t = fmaxf(t, __shfl_xor(t, off, 32));
      const float nm   = fmaxf(mrow[j], t);
      const float corr = exp2f((mrow[j] - nm) * KLS);
      const float e0   = exp2f((a0 - nm) * KLS);
      const float e1   = exp2f((a1 - nm) * KLS);
      float rs = e0 + e1;
#pragma unroll
      for (int off = 1; off < 16; off <<= 1) rs += __shfl_xor(rs, off, 32);
      lrow[j] = lrow[j] * corr + rs;
      mrow[j] = nm;
#pragma unroll
      for (int nc = 0; nc < 8; ++nc) o[nc][j] *= corr;
      p0[j] = e0; p1[j] = e1;
    }

#pragma unroll
    for (int j = 0; j < 8; ++j) {
      const int row = j + 8 * lh;
      lp[wvi][row][ln]      = (_Float16)p0[j];
      lp[wvi][row][16 + ln] = (_Float16)p1[j];
    }
    v16h pa = load_frag_h16(&lp[wvi][ln][lh * 16]);

#pragma unroll
    for (int nc = 0; nc < 8; ++nc) {
      v16h vb = load_frag_h16(vbase + (size_t)(nc * 16 + ln) * SEQ + k0 + lh * 16);
      o[nc] = wmma32(pa, vb, o[nc]);
    }
  }

#pragma unroll
  for (int j = 0; j < 8; ++j) {
    const float inv = 1.0f / lrow[j];
    const size_t rbase = (size_t)(b * SEQ + q0 + j + 8 * lh) * CDIM + h * HDIM;
#pragma unroll
    for (int nc = 0; nc < 8; ++nc)
      ah[rbase + nc * 16 + ln] = (_Float16)(o[nc][j] * inv);
  }
}

// ================= kernel 4: output projection (f16 x f16 -> f32) =================
__global__ __launch_bounds__(256)
void out_gemm_kernel(const _Float16* __restrict__ ah,
                     const _Float16* __restrict__ wo,
                     float* __restrict__ out)
{
  __shared__ _Float16 bt[2][64][32];

  const int lane = threadIdx.x & 31;
  const int wvi  = threadIdx.x >> 5;
  const int ln   = lane & 15;
  const int lh   = lane >> 4;
  const int rb   = blockIdx.x * 128 + wvi * 16;
  const int cb   = blockIdx.y * 64;

#if HAVE_TDM
  if (threadIdx.x < 32) {
    const unsigned int lds0 = (unsigned int)(uintptr_t)&bt[0][0][0];
    tdm_load_tile_2d(wo + (size_t)cb * CDIM, lds0, 32, 64, CDIM, CDIM, CDIM);
    __builtin_amdgcn_s_wait_tensorcnt(0);
  }
  __syncthreads();
#else
  coop_load_tile(bt[0], wo, cb, 0);
  __syncthreads();
#endif

  v8f acc[4] = {};
  const _Float16* arow = ah + (size_t)(rb + ln) * CDIM;

  for (int k0 = 0; k0 < CDIM; k0 += 32) {
    const int buf = (k0 >> 5) & 1;
    const bool more = (k0 + 32) < CDIM;
#if HAVE_TDM
    if (threadIdx.x < 32 && more) {
      const unsigned int lds1 = (unsigned int)(uintptr_t)&bt[buf ^ 1][0][0];
      tdm_load_tile_2d(wo + (size_t)cb * CDIM + (k0 + 32), lds1, 32, 64,
                       CDIM, CDIM, CDIM);
    }
#else
    __syncthreads();
    if (more) coop_load_tile(bt[buf ^ 1], wo, cb, k0 + 32);
#endif
    __builtin_prefetch(arow + k0 + 512, 0, 1);
    v16h a = load_frag_h16(arow + k0 + lh * 16);
#pragma unroll
    for (int nc = 0; nc < 4; ++nc) {
      v16h b = load_frag_h16(&bt[buf][nc * 16 + ln][lh * 16]);
      acc[nc] = wmma32(a, b, acc[nc]);
    }
#if HAVE_TDM
    if (threadIdx.x < 32 && more) __builtin_amdgcn_s_wait_tensorcnt(0);
#endif
    __syncthreads();
  }

#pragma unroll
  for (int nc = 0; nc < 4; ++nc)
#pragma unroll
    for (int j = 0; j < 8; ++j)
      out[(size_t)(rb + j + 8 * lh) * CDIM + cb + nc * 16 + ln] = acc[nc][j];
}

// ================= host launcher =================
extern "C" void kernel_launch(void* const* d_in, const int* in_sizes, int n_in,
                              void* d_out, int out_size, void* d_ws, size_t ws_size,
                              hipStream_t stream)
{
  const float* x  = (const float*)d_in[0];
  const float* Wq = (const float*)d_in[1];
  const float* Wk = (const float*)d_in[2];
  const float* Wv = (const float*)d_in[3];
  const float* Wo = (const float*)d_in[4];
  float* out = (float*)d_out;

  const size_t P  = (size_t)MTOT * CDIM;    // 8,388,608
  const size_t WN = (size_t)CDIM * CDIM;    // 4,194,304
  _Float16* qh  = (_Float16*)d_ws;          // [MTOT][CDIM]  (post-RoPE)
  _Float16* kh  = qh + P;                   // [MTOT][CDIM]  (post-RoPE)
  _Float16* vt  = qh + 2 * P;               // [B*H*HDIM][SEQ] (transposed V)
  _Float16* ah  = qh + 3 * P;               // [MTOT][CDIM]  (attention out)
  _Float16* xh  = qh + 4 * P;               // [MTOT][CDIM]  f16 input
  _Float16* wqf = qh + 5 * P;               // 4 x [CDIM][CDIM] f16 weights
  _Float16* wkf = wqf + WN;
  _Float16* wvf = wqf + 2 * WN;
  _Float16* wof = wqf + 3 * WN;
  (void)in_sizes; (void)n_in; (void)out_size; (void)ws_size;

  // kernel 0: one-time f32 -> f16 conversions
  cvt_f16_kernel<<<(int)(P / 8 / 256), 256, 0, stream>>>(x, xh, (int)(P / 8));
  cvt_f16_kernel<<<(int)(WN / 8 / 256), 256, 0, stream>>>(Wq, wqf, (int)(WN / 8));
  cvt_f16_kernel<<<(int)(WN / 8 / 256), 256, 0, stream>>>(Wk, wkf, (int)(WN / 8));
  cvt_f16_kernel<<<(int)(WN / 8 / 256), 256, 0, stream>>>(Wv, wvf, (int)(WN / 8));
  cvt_f16_kernel<<<(int)(WN / 8 / 256), 256, 0, stream>>>(Wo, wof, (int)(WN / 8));

  qkv_gemm_kernel<<<dim3(MTOT / 128, CDIM / 64, 3), 256, 0, stream>>>(
      xh, wqf, wkf, wvf, qh, kh, vt);

  rope_kernel<<<(MTOT * HEADS * 64) / 256, 256, 0, stream>>>(qh, kh);

  attn_kernel<<<dim3(SEQ / 128, BATCH * HEADS), 256, 0, stream>>>(qh, kh, vt, ah);

  out_gemm_kernel<<<dim3(MTOT / 128, CDIM / 64), 256, 0, stream>>>(ah, wof, out);
}
